// XConvLayer_PointCNN_21174188769385
// MI455X (gfx1250) — matrix operand order repaired and checked by reading, not compiled, not run on hardware
//
#include <hip/hip_runtime.h>
#include <hip/hip_bf16.h>

// ---------------------------------------------------------------------------
// PointCNN X-Conv for MI455X (gfx1250, wave32, WMMA).
// All four matmuls run through v_wmma_f32_16x16x32_f16 (f16 in, f32 acc).
// wl weight panel staged to LDS via global_load_async_to_lds_b128 (ASYNCcnt).
// ---------------------------------------------------------------------------

#define KNBR   16
#define NPTS   2048
#define NB     8
#define CIN    64
#define CLIFT  64
#define COUT   128
#define KD     32          // K*D
#define KM     256         // x_flat width (K*K)
#define HWID   512         // K * KD flattened
#define TWID   1024        // K * CLIFT flattened
#define BN_EPS 1e-3f

typedef __attribute__((ext_vector_type(16))) _Float16 v16h;
typedef __attribute__((ext_vector_type(8)))  _Float16 v8h;
typedef __attribute__((ext_vector_type(8)))  float    v8f;

union U16h { v16h v; v8h p[2]; _Float16 h[16]; };

__device__ __forceinline__ v8f zero8() {
  v8f z;
#pragma unroll
  for (int i = 0; i < 8; ++i) z[i] = 0.0f;
  return z;
}

// A fragment (16x32 f16, ISA 7.12.2): lane L holds row (L&15).
// halves e=0..7  -> K = kbase + aoff + e        (aoff = 0 for lanes 0-15, 8 for 16-31)
// halves e=8..15 -> K = kbase + 16 + aoff + (e-8)
// p0 must already point at rowbase + kbase + aoff.
__device__ __forceinline__ v16h load_a_frag(const _Float16* p0) {
  U16h u;
  u.p[0] = *(const v8h*)(p0);
  u.p[1] = *(const v8h*)(p0 + 16);
  return u.v;
}

// B fragment (32x16 f16): lane L holds column (L&15).
// halves e=0..7  -> K = kbase + koff + e        (koff = 0 for lanes 0-15, 8 for 16-31)
// halves e=8..15 -> K = kbase + 16 + koff + (e-8)
// p0 points at colbase + kbase + koff (K-contiguous transposed weights).
__device__ __forceinline__ v16h load_b_frag(const _Float16* p0) {
  U16h u;
  u.p[0] = *(const v8h*)(p0);
  u.p[1] = *(const v8h*)(p0 + 16);
  return u.v;
}

// ---------------------------------------------------------------------------
// Kernel 0: convert + transpose weights to f16, K-contiguous per output column
// ---------------------------------------------------------------------------
__global__ void convert_weights_kernel(const float* __restrict__ w2,
                                       const float* __restrict__ wl,
                                       const float* __restrict__ wf,
                                       _Float16* __restrict__ w2T,
                                       _Float16* __restrict__ wlT,
                                       _Float16* __restrict__ wfT) {
  int t = blockIdx.x * 256 + threadIdx.x;
  if (t < HWID * KM) {           // w2 flat [(k*32+d)][m] -> w2T[m][kd]
    int kd = t >> 8, m = t & 255;
    w2T[(size_t)m * HWID + kd] = (_Float16)w2[t];
  }
  if (t < TWID * COUT) {         // wf flat [(k*64+l)][o] -> wfT[o][kl]
    int kl = t >> 7, o = t & 127;
    wfT[(size_t)o * TWID + kl] = (_Float16)wf[t];
  }
  if (t < CIN * CLIFT) {         // wl [c][l] -> wlT[l][c]
    int c = t >> 6, l = t & 63;
    wlT[(size_t)l * CIN + c] = (_Float16)wl[t];
  }
}

// ---------------------------------------------------------------------------
// Kernel 1: brute-force KNN (top-16 smallest masked distance), idx out to ws.
// One thread per point; full batch xyz + validity staged in LDS.
// ---------------------------------------------------------------------------
__global__ void __launch_bounds__(128) knn_kernel(const float* __restrict__ xyz,
                                                  int* __restrict__ gidx) {
  __shared__ float sx[NPTS * 3];
  __shared__ float sm[NPTS];
  int b   = blockIdx.x >> 4;      // 16 blocks of 128 threads per batch
  int blk = blockIdx.x & 15;
  const float* base = xyz + (size_t)b * NPTS * 3;
  for (int t = threadIdx.x; t < NPTS; t += 128) {
    float x = base[t * 3 + 0], y = base[t * 3 + 1], z = base[t * 3 + 2];
    sx[t * 3 + 0] = x; sx[t * 3 + 1] = y; sx[t * 3 + 2] = z;
    sm[t] = (x != 0.0f || y != 0.0f || z != 0.0f) ? 1.0f : 0.0f;
  }
  __syncthreads();

  int n = blk * 128 + threadIdx.x;
  float cx = sx[n * 3 + 0], cy = sx[n * 3 + 1], cz = sx[n * 3 + 2];
  float cm = sm[n];

  float bd[KNBR]; int bi[KNBR];
#pragma unroll
  for (int i = 0; i < KNBR; ++i) { bd[i] = 3.0e38f; bi[i] = 0; }

  for (int j = 0; j < NPTS; ++j) {
    float dx = sx[j * 3 + 0] - cx;
    float dy = sx[j * 3 + 1] - cy;
    float dz = sx[j * 3 + 2] - cz;
    float d  = dx * dx + dy * dy + dz * dz;
    d = (cm * sm[j] != 0.0f) ? d : 1.0e9f;
    if (d < bd[KNBR - 1]) {
      bd[KNBR - 1] = d; bi[KNBR - 1] = j;
#pragma unroll
      for (int t = KNBR - 1; t > 0; --t) {
        if (bd[t] < bd[t - 1]) {
          float td = bd[t]; bd[t] = bd[t - 1]; bd[t - 1] = td;
          int   ti = bi[t]; bi[t] = bi[t - 1]; bi[t - 1] = ti;
        }
      }
    }
  }
  int* op = gidx + (((size_t)b * NPTS + n) << 4);
#pragma unroll
  for (int i = 0; i < KNBR; ++i) op[i] = bi[i];
}

// ---------------------------------------------------------------------------
// Kernel 2: fused X-Conv over a 16-point tile, 8 waves, 4 WMMA GEMM phases.
// ---------------------------------------------------------------------------
__global__ void __launch_bounds__(256)
xconv_kernel(const float* __restrict__ xyz, const float* __restrict__ feat,
             const float* __restrict__ w1,  const float* __restrict__ b1,
             const float* __restrict__ g1,  const float* __restrict__ be1,
             const float* __restrict__ m1,  const float* __restrict__ v1,
             const float* __restrict__ b2v,
             const float* __restrict__ gl,  const float* __restrict__ bel,
             const float* __restrict__ ml,  const float* __restrict__ vl,
             const float* __restrict__ gf,  const float* __restrict__ bef,
             const float* __restrict__ mf,  const float* __restrict__ vf,
             const int* __restrict__ gidx,
             const _Float16* __restrict__ w2T,
             const _Float16* __restrict__ wlT,
             const _Float16* __restrict__ wfT,
             float* __restrict__ out) {
  // sH doubles as: h matrix (phase 0 / GEMM1), then staged wl panel (GEMM2).
  __shared__ __align__(16) _Float16 sH[16 * HWID];   // 16KB
  __shared__ __align__(16) _Float16 sX[16 * KM];     // 8KB
  __shared__ __align__(16) _Float16 sL[256 * CLIFT]; // lifted/transformed 32KB
  __shared__ int   sIdx[256];
  __shared__ float sNm[256];
  __shared__ float sCm[16];

  const int tid  = threadIdx.x;
  const int lane = tid & 31;
  const int wv   = tid >> 5;            // wave 0..7
  const int b    = blockIdx.x >> 7;     // 128 tiles per batch
  const int n0   = (blockIdx.x & 127) << 4;

  const int aoff = (lane < 16) ? 0 : 8;   // A-fragment K sub-offset
  const int koff = (lane < 16) ? 0 : 8;   // B-fragment K sub-offset
  const int arow = lane & 15;             // A row / B column for this lane
  const int prow = (lane < 16) ? 0 : 8;   // C/D M base

  // Hint L2 about the streamed weight panels (global_prefetch_b8).
  __builtin_prefetch(w2T, 0, 1);
  __builtin_prefetch(wfT, 0, 1);

  // ---- Phase 0: gather neighbors, rel-xyz MLP (relu -> BN) into sH (f16) ----
  {
    int p  = tid >> 4;          // point in tile
    int kq = tid & 15;          // neighbor slot
    const float* cp = xyz + ((size_t)(b * NPTS) + n0 + p) * 3;
    float c0 = cp[0], c1 = cp[1], c2 = cp[2];
    float cm = (c0 != 0.0f || c1 != 0.0f || c2 != 0.0f) ? 1.0f : 0.0f;
    if (kq == 0) sCm[p] = cm;

    int nb = gidx[(((size_t)b * NPTS + n0 + p) << 4) + kq];
    sIdx[tid] = nb;
    const float* np = xyz + ((size_t)(b * NPTS) + nb) * 3;
    float x0 = np[0], x1 = np[1], x2 = np[2];
    float nm = (x0 != 0.0f || x1 != 0.0f || x2 != 0.0f) ? 1.0f : 0.0f;
    sNm[tid] = nm;
    float r0 = (x0 - c0) * nm, r1 = (x1 - c1) * nm, r2 = (x2 - c2) * nm;

    _Float16* hp = sH + p * HWID + kq * KD;
#pragma unroll
    for (int d = 0; d < KD; ++d) {
      float hv = r0 * w1[d] + r1 * w1[KD + d] + r2 * w1[2 * KD + d] + b1[d];
      hv = fmaxf(hv, 0.0f);                                             // relu
      hv = (hv - m1[d]) * g1[d] * __frsqrt_rn(v1[d] + BN_EPS) + be1[d]; // BN
      hp[d] = (_Float16)hv;
    }
  }
  __syncthreads();

  // ---- GEMM1: X[16 x 256] = H[16 x 512] * W2[512 x 256] (+b2) -> sX f16 ----
  {
#pragma unroll
    for (int t = 0; t < 2; ++t) {
      int nt = wv * 2 + t;                       // N tile 0..15
      v8f acc = zero8();
      const _Float16* arp  = sH + arow * HWID;
      const _Float16* brow = w2T + (size_t)(nt * 16 + arow) * HWID;
      for (int ks = 0; ks < 16; ++ks) {
        v16h A = load_a_frag(arp  + ks * 32 + aoff);
        v16h B = load_b_frag(brow + ks * 32 + koff);
        acc = __builtin_amdgcn_wmma_f32_16x16x32_f16(false, A, false, B,
                                                     (short)0, acc, false, false);
      }
      int m = nt * 16 + arow;
      float bias = b2v[m];
#pragma unroll
      for (int r = 0; r < 8; ++r)
        sX[(prow + r) * KM + m] = (_Float16)(acc[r] + bias);
    }
  }
  __syncthreads();   // sH (h matrix) is dead past this point

  // ---- Async-stage wl panel (8KB) into the sH overlay via the async engine.
  // Each of 256 threads moves 2 x 16B chunks (offset applies to both LDS and global address)
  {
    unsigned lbase = (unsigned)(size_t)(__attribute__((address_space(3))) char*)(void*)sH;
    unsigned o0 = (unsigned)tid * 32u;
    asm volatile("global_load_async_to_lds_b128 %0, %1, %2"
                 :: "v"(lbase + o0), "v"(o0), "s"(wlT) : "memory");
    asm volatile("global_load_async_to_lds_b128 %0, %1, %2 offset:16"
                 :: "v"(lbase + o0), "v"(o0), "s"(wlT) : "memory");
    asm volatile("s_wait_asynccnt 0x0" ::: "memory");
  }
  __syncthreads();   // all waves' async chunks landed

  // ---- GEMM2: lifted[256 x 64] = NF[256 x 64] * Wl[64 x 64]; BN->relu->mask --
  {
    const _Float16* sWl = sH;   // staged wl panel, layout [l][c]
#pragma unroll
    for (int t = 0; t < 2; ++t) {
      int mt   = wv * 2 + t;                     // M tile over 256 (pt,nbr) rows
      int grow = mt * 16 + arow;                 // this lane's A row
      const float* fp = feat + ((size_t)(b * NPTS) + sIdx[grow]) * CIN;
      v8f acc[4] = {zero8(), zero8(), zero8(), zero8()};
#pragma unroll
      for (int ks = 0; ks < 2; ++ks) {
        U16h a;
        const float* s0 = fp + ks * 32 + aoff;
#pragma unroll
        for (int e = 0; e < 8; ++e) a.h[e]     = (_Float16)s0[e];
#pragma unroll
        for (int e = 0; e < 8; ++e) a.h[8 + e] = (_Float16)s0[16 + e];
#pragma unroll
        for (int nt = 0; nt < 4; ++nt) {
          v16h B = load_b_frag(sWl + (size_t)(nt * 16 + arow) * CIN + ks * 32 + koff);
          acc[nt] = __builtin_amdgcn_wmma_f32_16x16x32_f16(false, a.v, false, B,
                                                           (short)0, acc[nt], false, false);
        }
      }
      int rbase = mt * 16 + prow;
#pragma unroll
      for (int nt = 0; nt < 4; ++nt) {
        int l = nt * 16 + arow;
        float scale = gl[l] * __frsqrt_rn(vl[l] + BN_EPS);
        float mm = ml[l], be = bel[l];
#pragma unroll
        for (int r = 0; r < 8; ++r) {
          int row = rbase + r;
          float x = (acc[nt][r] - mm) * scale + be;   // BN
          x = fmaxf(x, 0.0f) * sNm[row];              // relu + mask
          sL[row * CLIFT + l] = (_Float16)x;
        }
      }
    }
  }
  __syncthreads();

  // ---- GEMM3: per point, T[16 x 64] = X[16 x 16 (pad 32)] * lifted[16 x 64] --
  {
#pragma unroll
    for (int t = 0; t < 2; ++t) {
      int p = wv * 2 + t;
      U16h a;                                     // A: X padded to 16x32
      a.p[0] = *(const v8h*)(sX + p * KM + arow * 16 + aoff);
#pragma unroll
      for (int e = 0; e < 8; ++e) a.h[8 + e] = (_Float16)0.0f;  // K=16..31 pad

      v8f acc[4];
#pragma unroll
      for (int nt = 0; nt < 4; ++nt) {
        U16h bf;                                  // B: lifted rows, K>=16 pad
#pragma unroll
        for (int e = 0; e < 16; ++e) {
          int k = (e < 8) ? koff + e : 16 + koff + (e - 8);
          bf.h[e] = (k < 16) ? sL[(p * 16 + k) * CLIFT + nt * 16 + arow]
                             : (_Float16)0.0f;
        }
        acc[nt] = __builtin_amdgcn_wmma_f32_16x16x32_f16(false, a.v, false, bf.v,
                                                         (short)0, zero8(), false, false);
      }
      // Overwrite lifted[p] with transformed[p] (only this wave touches p).
#pragma unroll
      for (int nt = 0; nt < 4; ++nt)
#pragma unroll
        for (int r = 0; r < 8; ++r)
          sL[(p * 16 + prow + r) * CLIFT + nt * 16 + arow] = (_Float16)acc[nt][r];
    }
  }
  __syncthreads();

  // ---- GEMM4: out[16 x 128] = T[16 x 1024] * Wf[1024 x 128]; BN->relu->mask --
  {
    int nt = wv;                                  // 8 N tiles of 16 cols
    v8f acc = zero8();
    const _Float16* arp  = sL + arow * TWID;      // row = point (lane&15)
    const _Float16* brow = wfT + (size_t)(nt * 16 + arow) * TWID;
    for (int ks = 0; ks < 32; ++ks) {
      v16h A = load_a_frag(arp  + ks * 32 + aoff);
      v16h B = load_b_frag(brow + ks * 32 + koff);
      acc = __builtin_amdgcn_wmma_f32_16x16x32_f16(false, A, false, B,
                                                   (short)0, acc, false, false);
    }
    int o = nt * 16 + arow;
    float scale = gf[o] * __frsqrt_rn(vf[o] + BN_EPS);
    float mm = mf[o], be = bef[o];
#pragma unroll
    for (int r = 0; r < 8; ++r) {
      int p = prow + r;
      float x = (acc[r] - mm) * scale + be;
      x = fmaxf(x, 0.0f) * sCm[p];
      out[((size_t)(b * NPTS) + n0 + p) * COUT + o] = x;
    }
  }
}

// ---------------------------------------------------------------------------
extern "C" void kernel_launch(void* const* d_in, const int* in_sizes, int n_in,
                              void* d_out, int out_size, void* d_ws, size_t ws_size,
                              hipStream_t stream) {
  (void)in_sizes; (void)n_in; (void)out_size; (void)ws_size;
  const float* xyz  = (const float*)d_in[0];
  const float* feat = (const float*)d_in[1];
  const float* w1   = (const float*)d_in[2];
  const float* b1   = (const float*)d_in[3];
  const float* g1   = (const float*)d_in[4];
  const float* be1  = (const float*)d_in[5];
  const float* m1   = (const float*)d_in[6];
  const float* v1   = (const float*)d_in[7];
  const float* w2   = (const float*)d_in[8];
  const float* b2   = (const float*)d_in[9];
  const float* wl   = (const float*)d_in[10];
  const float* gl   = (const float*)d_in[11];
  const float* bel  = (const float*)d_in[12];
  const float* ml   = (const float*)d_in[13];
  const float* vl   = (const float*)d_in[14];
  const float* wf   = (const float*)d_in[15];
  const float* gf   = (const float*)d_in[16];
  const float* bef  = (const float*)d_in[17];
  const float* mf   = (const float*)d_in[18];
  const float* vf   = (const float*)d_in[19];

  // Workspace layout (all 16B aligned):
  //   [0, 2 MiB)        idx   int  [B][N][16]
  //   [+,   256 KiB)    w2T   f16  [256][512]
  //   [+,     8 KiB)    wlT   f16  [64][64]
  //   [+,   256 KiB)    wfT   f16  [128][1024]
  char* ws = (char*)d_ws;
  int*      idx = (int*)ws;
  _Float16* w2T = (_Float16*)(ws + (size_t)2097152);
  _Float16* wlT = (_Float16*)(ws + (size_t)2097152 + 262144);
  _Float16* wfT = (_Float16*)(ws + (size_t)2097152 + 262144 + 8192);

  convert_weights_kernel<<<512, 256, 0, stream>>>(w2, wl, wf, w2T, wlT, wfT);
  knn_kernel<<<NB * (NPTS / 128), 128, 0, stream>>>(xyz, idx);
  xconv_kernel<<<NB * (NPTS / 16), 256, 0, stream>>>(
      xyz, feat, w1, b1, g1, be1, m1, v1, b2, gl, bel, ml, vl,
      gf, bef, mf, vf, idx, w2T, wlT, wfT, (float*)d_out);
}
